// TwoStage3DDetector_24541443129446
// MI455X (gfx1250) — matrix operand order repaired and checked by reading, not compiled, not run on hardware
//
#include <hip/hip_runtime.h>

typedef _Float16 h16;
typedef __attribute__((ext_vector_type(16))) _Float16 v16h;
typedef __attribute__((ext_vector_type(8)))  _Float16 v8h;
typedef __attribute__((ext_vector_type(8)))  float    v8f;

// ---------------------------------------------------------------------------
// Problem constants (match reference shapes)
// ---------------------------------------------------------------------------
#define B_    16
#define C_    2048
#define HF    20
#define WF    60
#define NBOX  256
#define MROWS (B_ * NBOX)     // 4096
#define KCOMB 2176            // roi(2048) + intr(128)
#define IIH   (HF + 1)        // 21
#define IIW   (WF + 1)        // 61

#if __has_builtin(__builtin_amdgcn_s_wait_asynccnt)
#define WAIT_ASYNC0() __builtin_amdgcn_s_wait_asynccnt(0)
#else
#define WAIT_ASYNC0() asm volatile("s_wait_asynccnt 0x0" ::: "memory")
#endif

// ---------------------------------------------------------------------------
// 1) Intrinsics MLP: [16,4] -> 64 (ReLU) -> 128 (ReLU), f16 output
// ---------------------------------------------------------------------------
__global__ __launch_bounds__(128)
void intr_kernel(const float* __restrict__ K9,
                 const float* __restrict__ W0, const float* __restrict__ b0,
                 const float* __restrict__ W1, const float* __restrict__ b1,
                 h16* __restrict__ intrF)
{
    __shared__ float h1[16][64];
    const int t = threadIdx.x;
    for (int i = t; i < 16 * 64; i += 128) {
        const int b = i >> 6, j = i & 63;
        const float* kk = K9 + b * 9;
        const float fx = kk[0], fy = kk[4], cx = kk[2], cy = kk[5];
        float s = fx * W0[0 * 64 + j] + fy * W0[1 * 64 + j]
                + cx * W0[2 * 64 + j] + cy * W0[3 * 64 + j] + b0[j];
        h1[b][j] = fmaxf(s, 0.0f);
    }
    __syncthreads();
    for (int i = t; i < 16 * 128; i += 128) {
        const int b = i >> 7, j = i & 127;
        float s = b1[j];
        for (int k = 0; k < 64; ++k) s += h1[b][k] * W1[k * 128 + j];
        intrF[i] = (h16)fmaxf(s, 0.0f);
    }
}

// ---------------------------------------------------------------------------
// 2) Integral image, transposed to channel-innermost: ii[b][y][x][c] (f32)
// ---------------------------------------------------------------------------
__global__ __launch_bounds__(256)
void integral_kernel(const float* __restrict__ feat, float* __restrict__ ii)
{
    __shared__ float sm[8][HF][IIW];
    const int lane  = threadIdx.x & 31;
    const int w     = threadIdx.x >> 5;
    const int plane = blockIdx.x * 8 + w;           // 0..32767
    const int b     = plane >> 11;
    const int c     = plane & (C_ - 1);

    const float* src = feat + (size_t)plane * (HF * WF);
    for (int i = lane; i < HF * WF; i += 32) {
        const int y = i / WF, x = i - y * WF;
        sm[w][y][x + 1] = src[i];
    }
    if (lane < HF) sm[w][lane][0] = 0.0f;
    __syncthreads();

    if (lane < HF) {                                 // row prefix
        float run = 0.0f;
        for (int x = 1; x <= WF; ++x) { run += sm[w][lane][x]; sm[w][lane][x] = run; }
    }
    __syncthreads();

    for (int xx = lane; xx < IIW; xx += 32) {        // column prefix + store
        float run = 0.0f;
        ii[(((size_t)b * IIH + 0) * IIW + xx) * C_ + c] = 0.0f;
        for (int y = 1; y <= HF; ++y) {
            run += sm[w][y - 1][xx];
            ii[(((size_t)b * IIH + y) * IIW + xx) * C_ + c] = run;
        }
    }
}

// ---------------------------------------------------------------------------
// 3) ROI gather (4 corners, contiguous along c) + append intr -> combined f16
// ---------------------------------------------------------------------------
__global__ __launch_bounds__(256)
void roi_kernel(const float* __restrict__ ii, const float* __restrict__ boxes,
                const h16* __restrict__ intrF, h16* __restrict__ comb,
                const int* __restrict__ img_h_p, const int* __restrict__ img_w_p)
{
    const int idx = blockIdx.x;                      // 0..4095
    const int b = idx >> 8;
    const int n = idx & (NBOX - 1);
    const float img_w = (float)img_w_p[0];
    const float img_h = (float)img_h_p[0];

    const float* bx = boxes + ((size_t)b * NBOX + n) * 4;
    const float x1 = bx[0], y1 = bx[1], x2 = bx[2], y2 = bx[3];
    const int x1n = min(max((int)(x1 / img_w * (float)WF), 0), WF - 1);
    const int x2n = min(max((int)(x2 / img_w * (float)WF), 0), WF - 1);
    const int y1n = min(max((int)(y1 / img_h * (float)HF), 0), HF - 1);
    const int y2n = min(max((int)(y2 / img_h * (float)HF), 0), HF - 1);

    const size_t base = (size_t)b * IIH * IIW * C_;
    const size_t o22 = base + ((size_t)(y2n + 1) * IIW + (x2n + 1)) * C_;
    const size_t o12 = base + ((size_t)(y1n)     * IIW + (x2n + 1)) * C_;
    const size_t o21 = base + ((size_t)(y2n + 1) * IIW + (x1n))     * C_;
    const size_t o11 = base + ((size_t)(y1n)     * IIW + (x1n))     * C_;
    const float inv_area = 1.0f / (float)((y2n - y1n + 1) * (x2n - x1n + 1));

    h16* row = comb + (size_t)idx * KCOMB;
    for (int c = threadIdx.x; c < C_; c += 256) {
        const float s = ii[o22 + c] - ii[o12 + c] - ii[o21 + c] + ii[o11 + c];
        row[c] = (h16)(s * inv_area);
    }
    if (threadIdx.x < 128)
        row[C_ + threadIdx.x] = intrF[b * 128 + threadIdx.x];
}

// ---------------------------------------------------------------------------
// 4) Weight prepack: f32 W[K][N] -> f16 fragment-native layout
//    Wp[((nt*kT + kt)*32 + lane)*16 + e] = W[kt*32 + (lane>>4)*16 + e][nt*16 + (lane&15)]
// ---------------------------------------------------------------------------
__global__ __launch_bounds__(256)
void wprep_kernel(const float* __restrict__ W, h16* __restrict__ Wp, int K, int N)
{
    const int kT = K >> 5;
    const size_t total = (size_t)(N >> 4) * kT * 512;
    for (size_t idx = (size_t)blockIdx.x * 256 + threadIdx.x; idx < total;
         idx += (size_t)gridDim.x * 256) {
        const int    v    = (int)(idx & 511);
        const size_t tile = idx >> 9;
        const int    nt   = (int)(tile / kT);
        const int    kt   = (int)(tile % kT);
        const int    lane = v >> 4, e = v & 15;
        const int    nn   = nt * 16 + (lane & 15);
        const int    kk   = kt * 32 + ((lane >> 4) << 4) + e;
        Wp[idx] = (h16)W[(size_t)kk * N + nn];
    }
}

// ---------------------------------------------------------------------------
// 5) WMMA GEMM: Out[M,N] = act(A[M,K] @ W + bias)
//    Wave tile 16x128 (8 accumulators), block tile 128x128 (8 waves).
//    B tile staged in LDS via GLOBAL_LOAD_ASYNC_TO_LDS (double-buffered);
//    A fragment software-pipelined one K-step ahead in registers;
//    B fragments flow through a 3-deep rotating register pipeline so each
//    WMMA's operand was ds-loaded 3 WMMA-issues earlier (partial dscnt waits).
// ---------------------------------------------------------------------------
__global__ __launch_bounds__(256)
void gemm_wmma_kernel(const h16* __restrict__ A, int lda,
                      const h16* __restrict__ Wp,
                      const float* __restrict__ bias,
                      h16* __restrict__ Out,
                      int K, int N, int relu)
{
    __shared__ __align__(32) h16 smB[2][8 * 512];   // 2 buffers x 8 KB

    const int t    = threadIdx.x;
    const int lane = t & 31;
    const int wave = t >> 5;
    const int m0   = blockIdx.y * 128 + wave * 16;
    const int n0   = blockIdx.x * 128;
    const int kT   = K >> 5;
    const int hs   = lane >> 4;          // half-wave select
    const int lr   = lane & 15;

    const h16* aRow = A + (size_t)(m0 + lr) * lda + hs * 16;

    // A fragment loader (contiguous 16-bit A layout: one 32-B run per lane, K + 16*hs + 0..15)
    auto loadA = [&](int kt) -> v16h {
        const v8h* ap  = (const v8h*)(aRow + (size_t)kt * 32);
        const v8h  alo = ap[0];          // K + 16*hs + {0..7}
        const v8h  ahi = ap[1];          // K + 16*hs + {8..15}
        v16h a;
#pragma unroll
        for (int i = 0; i < 8; ++i) { a[i] = alo[i]; a[i + 8] = ahi[i]; }
        return a;
    };

    // B staging: thread t copies 32 B of subtile bj=(t>>5) per K-step.
    const int  bj  = t >> 5;             // 0..7 (n-subtile)
    const int  bs  = t & 31;             // lane slot within subtile
    const h16* gB0 = Wp + (size_t)((n0 >> 4) + bj) * kT * 512 + bs * 16;

    auto issueB = [&](int kt, int bb) {
        const h16* g = gB0 + (size_t)kt * 512;
        unsigned l = (unsigned)(uintptr_t)(&smB[bb][bj * 512 + bs * 16]);
        unsigned long long gv = (unsigned long long)(uintptr_t)g;
        asm volatile("global_load_async_to_lds_b128 %0, %1, off"
                     :: "v"(l), "v"(gv) : "memory");
        asm volatile("global_load_async_to_lds_b128 %0, %1, off offset:16"
                     :: "v"(l), "v"(gv) : "memory");
    };

    v8f acc[8];
#pragma unroll
    for (int j = 0; j < 8; ++j) acc[j] = (v8f){};

    issueB(0, 0);
    v16h aNext = loadA(0);               // pipelined A fragment

    for (int kt = 0; kt < kT; ++kt) {
        const int bb = kt & 1;
        WAIT_ASYNC0();          // own async loads for buffer bb complete
        __syncthreads();        // all waves' loads landed; prior reads retired

        if (kt + 1 < kT) issueB(kt + 1, bb ^ 1);

        const v16h aCur = aNext;
        const int  ktn  = (kt + 1 < kT) ? kt + 1 : kt;   // clamped (no OOB)
        aNext = loadA(ktn);

        // 3-deep rotating B-fragment pipeline: operand of WMMA j was loaded
        // 3 WMMA-issues earlier; DS returns in order -> partial dscnt waits.
        const h16* sb = &smB[bb][lane * 16];
#define LDB(j) (*(const v16h*)(sb + (j) * 512))
        v16h f0 = LDB(0);
        v16h f1 = LDB(1);
        v16h f2 = LDB(2);
        acc[0] = __builtin_amdgcn_wmma_f32_16x16x32_f16(false, aCur, false, f0, (short)0, acc[0], false, false);
        f0 = LDB(3);
        acc[1] = __builtin_amdgcn_wmma_f32_16x16x32_f16(false, aCur, false, f1, (short)0, acc[1], false, false);
        f1 = LDB(4);
        acc[2] = __builtin_amdgcn_wmma_f32_16x16x32_f16(false, aCur, false, f2, (short)0, acc[2], false, false);
        f2 = LDB(5);
        acc[3] = __builtin_amdgcn_wmma_f32_16x16x32_f16(false, aCur, false, f0, (short)0, acc[3], false, false);
        f0 = LDB(6);
        acc[4] = __builtin_amdgcn_wmma_f32_16x16x32_f16(false, aCur, false, f1, (short)0, acc[4], false, false);
        f1 = LDB(7);
        acc[5] = __builtin_amdgcn_wmma_f32_16x16x32_f16(false, aCur, false, f2, (short)0, acc[5], false, false);
        acc[6] = __builtin_amdgcn_wmma_f32_16x16x32_f16(false, aCur, false, f0, (short)0, acc[6], false, false);
        acc[7] = __builtin_amdgcn_wmma_f32_16x16x32_f16(false, aCur, false, f1, (short)0, acc[7], false, false);
#undef LDB
    }

#pragma unroll
    for (int j = 0; j < 8; ++j) {
        const int   nj = n0 + 16 * j + lr;
        const float bv = bias[nj];
#pragma unroll
        for (int r = 0; r < 8; ++r) {
            const int m = m0 + hs * 8 + r;     // C/D layout: VGPR r -> M=r (+8 hi half)
            float v = acc[j][r] + bv;
            if (relu) v = fmaxf(v, 0.0f);
            Out[(size_t)m * N + nj] = (h16)v;
        }
    }
}

// ---------------------------------------------------------------------------
// 6) Head GEMV (N<=24): wave per row, shuffle reduce; act 0=none, 1=sigmoid
// ---------------------------------------------------------------------------
template <int NOUT>
__global__ __launch_bounds__(256)
void head_kernel(const h16* __restrict__ X, const float* __restrict__ W,
                 const float* __restrict__ bias, float* __restrict__ out,
                 int K, int act)
{
    const int lane = threadIdx.x & 31;
    const int row  = blockIdx.x * 8 + (threadIdx.x >> 5);

    float acc[NOUT];
#pragma unroll
    for (int n = 0; n < NOUT; ++n) acc[n] = 0.0f;

    const h16* xr = X + (size_t)row * K;
    for (int k = lane; k < K; k += 32) {
        const float xv = (float)xr[k];
        const float* wr = W + (size_t)k * NOUT;
#pragma unroll
        for (int n = 0; n < NOUT; ++n) acc[n] += xv * wr[n];
    }
#pragma unroll
    for (int n = 0; n < NOUT; ++n) {
        float v = acc[n];
        for (int off = 16; off; off >>= 1) v += __shfl_xor(v, off, 32);
        if (lane == 0) {
            v += bias[n];
            if (act == 1) v = 1.0f / (1.0f + expf(-v));
            out[(size_t)row * NOUT + n] = v;
        }
    }
}

// ---------------------------------------------------------------------------
// Host launcher
// ---------------------------------------------------------------------------
extern "C" void kernel_launch(void* const* d_in, const int* in_sizes, int n_in,
                              void* d_out, int out_size, void* d_ws, size_t ws_size,
                              hipStream_t stream)
{
    (void)in_sizes; (void)n_in; (void)out_size; (void)ws_size;

    const float* feats  = (const float*)d_in[0];
    const float* Kmat   = (const float*)d_in[1];
    const float* boxes  = (const float*)d_in[2];
    const float* intrW0 = (const float*)d_in[3];
    const float* intrB0 = (const float*)d_in[4];
    const float* intrW1 = (const float*)d_in[5];
    const float* intrB1 = (const float*)d_in[6];
    const float* depW0  = (const float*)d_in[7];
    const float* depB0  = (const float*)d_in[8];
    const float* depW1  = (const float*)d_in[9];
    const float* depB1  = (const float*)d_in[10];
    const float* depW2  = (const float*)d_in[11];
    const float* depB2  = (const float*)d_in[12];
    const float* dimW0  = (const float*)d_in[13];
    const float* dimB0  = (const float*)d_in[14];
    const float* dimW1  = (const float*)d_in[15];
    const float* dimB1  = (const float*)d_in[16];
    const float* rbW0   = (const float*)d_in[17];
    const float* rbB0   = (const float*)d_in[18];
    const float* rbW1   = (const float*)d_in[19];
    const float* rbB1   = (const float*)d_in[20];
    const float* rrW0   = (const float*)d_in[21];
    const float* rrB0   = (const float*)d_in[22];
    const float* rrW1   = (const float*)d_in[23];
    const float* rrB1   = (const float*)d_in[24];
    const float* iouW0  = (const float*)d_in[25];
    const float* iouB0  = (const float*)d_in[26];
    const float* iouW1  = (const float*)d_in[27];
    const float* iouB1  = (const float*)d_in[28];
    const float* iouW2  = (const float*)d_in[29];
    const float* iouB2  = (const float*)d_in[30];
    const float* fgW0   = (const float*)d_in[31];
    const float* fgB0   = (const float*)d_in[32];
    const float* fgW1   = (const float*)d_in[33];
    const float* fgB1   = (const float*)d_in[34];
    const int*   imgh   = (const int*)d_in[35];
    const int*   imgw   = (const int*)d_in[36];

    char* ws = (char*)d_ws;
    size_t off = 0;
    auto carve = [&](size_t bytes) -> void* {
        void* p = ws + off;
        off = (off + bytes + 255) & ~(size_t)255;
        return p;
    };
    float* ii    = (float*)carve((size_t)B_ * IIH * IIW * C_ * 4);    // ~160 MB
    h16*   comb  = (h16*)  carve((size_t)MROWS * KCOMB * 2);
    h16*   intrF = (h16*)  carve((size_t)B_ * 128 * 2);
    h16*   hD1   = (h16*)  carve((size_t)MROWS * 512 * 2);
    h16*   hD2   = (h16*)  carve((size_t)MROWS * 256 * 2);
    h16*   hDim1 = (h16*)  carve((size_t)MROWS * 512 * 2);
    h16*   hRB   = (h16*)  carve((size_t)MROWS * 256 * 2);
    h16*   hRR   = (h16*)  carve((size_t)MROWS * 256 * 2);
    h16*   hI1   = (h16*)  carve((size_t)MROWS * 512 * 2);
    h16*   hI2   = (h16*)  carve((size_t)MROWS * 256 * 2);
    h16*   hFG   = (h16*)  carve((size_t)MROWS * 256 * 2);
    h16*   pDep0 = (h16*)  carve((size_t)KCOMB * 512 * 2);
    h16*   pDep1 = (h16*)  carve((size_t)512 * 256 * 2);
    h16*   pDim0 = (h16*)  carve((size_t)KCOMB * 512 * 2);
    h16*   pRB0  = (h16*)  carve((size_t)C_ * 256 * 2);
    h16*   pRR0  = (h16*)  carve((size_t)C_ * 256 * 2);
    h16*   pIou0 = (h16*)  carve((size_t)C_ * 512 * 2);
    h16*   pIou1 = (h16*)  carve((size_t)512 * 256 * 2);
    h16*   pFG0  = (h16*)  carve((size_t)C_ * 256 * 2);

    // --- Stage 1: intrinsics MLP + integral image + ROI build ---
    intr_kernel<<<1, 128, 0, stream>>>(Kmat, intrW0, intrB0, intrW1, intrB1, intrF);
    integral_kernel<<<(B_ * C_) / 8, 256, 0, stream>>>(feats, ii);
    roi_kernel<<<MROWS, 256, 0, stream>>>(ii, boxes, intrF, comb, imgh, imgw);

    // --- Stage 2: weight prepack into WMMA B-fragment layout ---
    auto prep = [&](const float* W, h16* Wp, int K, int N) {
        size_t total = (size_t)(N / 16) * (K / 32) * 512;
        int blocks = (int)((total + 255) / 256);
        wprep_kernel<<<blocks, 256, 0, stream>>>(W, Wp, K, N);
    };
    prep(depW0, pDep0, KCOMB, 512);
    prep(depW1, pDep1, 512, 256);
    prep(dimW0, pDim0, KCOMB, 512);
    prep(rbW0,  pRB0,  C_, 256);
    prep(rrW0,  pRR0,  C_, 256);
    prep(iouW0, pIou0, C_, 512);
    prep(iouW1, pIou1, 512, 256);
    prep(fgW0,  pFG0,  C_, 256);

    // --- Stage 3: hidden GEMMs (WMMA + async-LDS double buffering, ReLU) ---
    auto gemm = [&](const h16* A, int lda, const h16* Wp, const float* bias,
                    h16* Out, int K, int N) {
        dim3 grid(N / 128, MROWS / 128);
        gemm_wmma_kernel<<<grid, 256, 0, stream>>>(A, lda, Wp, bias, Out, K, N, 1);
    };
    gemm(comb, KCOMB, pDep0, depB0, hD1,   KCOMB, 512);   // depth L1
    gemm(hD1,  512,   pDep1, depB1, hD2,   512,   256);   // depth L2
    gemm(comb, KCOMB, pDim0, dimB0, hDim1, KCOMB, 512);   // dim   L1
    gemm(comb, KCOMB, pRB0,  rbB0,  hRB,   C_,    256);   // rot_bin L1 (roi only)
    gemm(comb, KCOMB, pRR0,  rrB0,  hRR,   C_,    256);   // rot_res L1
    gemm(comb, KCOMB, pIou0, iouB0, hI1,   C_,    512);   // iou   L1
    gemm(hI1,  512,   pIou1, iouB1, hI2,   512,   256);   // iou   L2
    gemm(comb, KCOMB, pFG0,  fgB0,  hFG,   C_,    256);   // fg    L1

    // --- Stage 4: heads (GEMV), outputs concatenated in return order ---
    float* out = (float*)d_out;
    const int HB = MROWS / 8;   // 512 blocks, wave per row
    head_kernel<3> <<<HB, 256, 0, stream>>>(hD2,   depW2, depB2, out + 0,      256, 0);
    head_kernel<3> <<<HB, 256, 0, stream>>>(hDim1, dimW1, dimB1, out + 12288,  512, 0);
    head_kernel<24><<<HB, 256, 0, stream>>>(hRB,   rbW1,  rbB1,  out + 24576,  256, 0);
    head_kernel<24><<<HB, 256, 0, stream>>>(hRR,   rrW1,  rrB1,  out + 122880, 256, 0);
    head_kernel<1> <<<HB, 256, 0, stream>>>(hI2,   iouW2, iouB2, out + 221184, 256, 1);
    head_kernel<2> <<<HB, 256, 0, stream>>>(hFG,   fgW1,  fgB1,  out + 225280, 256, 0);
}
